// HeteroAttnGNN_36490042146918
// MI455X (gfx1250) — matrix operand order repaired and checked by reading, not compiled, not run on hardware
//
#include <hip/hip_runtime.h>
#include <math.h>

// ---------------- constants (match reference) ----------------
static const int kNF  = 50000;
static const int kNC  = 5000;
static const int kE   = 200000;
static const int kB   = 128;

// ---------------- WMMA types ----------------
typedef __attribute__((ext_vector_type(16))) __bf16 v16bf;
typedef __attribute__((ext_vector_type(2)))  __bf16 v2bf;
typedef __attribute__((ext_vector_type(8)))  float  v8f;

__device__ inline __bf16 us2bf(unsigned short h) {
  __bf16 r; __builtin_memcpy(&r, &h, 2); return r;
}
// pairwise f32 -> bf16; prefer the packed convert builtin, else fptrunc
// (native bf16 convert on gfx1250), never the manual bit-twiddle.
__device__ inline void cvt2bf(float x, float y, v16bf& v, int base) {
#if __has_builtin(__builtin_amdgcn_cvt_pk_bf16_f32)
  v2bf p = __builtin_amdgcn_cvt_pk_bf16_f32(x, y);
  v[base] = p[0]; v[base + 1] = p[1];
#else
  v[base] = (__bf16)x; v[base + 1] = (__bf16)y;
#endif
}
__device__ inline v16bf load16bf(const unsigned short* p) {
  // 16 contiguous bf16 (32B, aligned) -> fragment elements 0..15
  uint4 r0 = *(const uint4*)(p);
  uint4 r1 = *(const uint4*)(p + 8);
  v16bf v;
  v[0]=us2bf((unsigned short)r0.x);  v[1]=us2bf((unsigned short)(r0.x>>16));
  v[2]=us2bf((unsigned short)r0.y);  v[3]=us2bf((unsigned short)(r0.y>>16));
  v[4]=us2bf((unsigned short)r0.z);  v[5]=us2bf((unsigned short)(r0.z>>16));
  v[6]=us2bf((unsigned short)r0.w);  v[7]=us2bf((unsigned short)(r0.w>>16));
  v[8]=us2bf((unsigned short)r1.x);  v[9]=us2bf((unsigned short)(r1.x>>16));
  v[10]=us2bf((unsigned short)r1.y); v[11]=us2bf((unsigned short)(r1.y>>16));
  v[12]=us2bf((unsigned short)r1.z); v[13]=us2bf((unsigned short)(r1.z>>16));
  v[14]=us2bf((unsigned short)r1.w); v[15]=us2bf((unsigned short)(r1.w>>16));
  return v;
}

__device__ inline void atomicMaxF(float* addr, float val) {
  if (val >= 0.0f) atomicMax((int*)addr, __float_as_int(val));
  else             atomicMin((unsigned int*)addr, __float_as_uint(val));
}

// ---------------- GEMM: C[M,N] = act(A[M,K](f32) * Bt[N,K](bf16)^T + bias) ----------------
// One wave -> 16x128 tile (8 accumulators). K % 32 == 0, N % 128 == 0.
// Edge rows load from a clamped pointer (their results are masked at the store),
// so the hot loop has no EXEC juggling. No manual unroll: keeps register
// pressure low so the scheduler double-buffers B loads (s_wait_loadcnt 0x2).
__global__ __launch_bounds__(128) void k_gemm_bf16(
    const float* __restrict__ A, const unsigned short* __restrict__ Bt,
    const float* __restrict__ bias, float* __restrict__ Cm,
    int M, int N, int K, int relu)
{
  const int wave    = blockIdx.x * 4 + (threadIdx.x >> 5);
  const int lane    = threadIdx.x & 31;
  const int nt8     = N >> 7;                       // tiles of 128 columns
  const int mt      = wave / nt8;
  if (mt * 16 >= M) return;                         // uniform per-wave exit
  const int nt      = (wave % nt8) << 7;
  const int lanelow = lane & 15;
  const int half    = lane >> 4;
  const int mrow    = mt * 16 + lanelow;
  // Clamped source row: out-of-range lanes read row 0; their outputs are
  // never stored (D row m depends only on A row m).
  const float* arow = A + (size_t)(mrow < M ? mrow : 0) * K + half * 16;
  // B k order chosen here (same as A): lane half holds contiguous K run [half*16, half*16+16)
  const unsigned short* bp0 = Bt + (size_t)(nt + lanelow      ) * K + half * 16;
  const unsigned short* bp1 = Bt + (size_t)(nt + lanelow + 16 ) * K + half * 16;
  const unsigned short* bp2 = Bt + (size_t)(nt + lanelow + 32 ) * K + half * 16;
  const unsigned short* bp3 = Bt + (size_t)(nt + lanelow + 48 ) * K + half * 16;
  const unsigned short* bp4 = Bt + (size_t)(nt + lanelow + 64 ) * K + half * 16;
  const unsigned short* bp5 = Bt + (size_t)(nt + lanelow + 80 ) * K + half * 16;
  const unsigned short* bp6 = Bt + (size_t)(nt + lanelow + 96 ) * K + half * 16;
  const unsigned short* bp7 = Bt + (size_t)(nt + lanelow + 112) * K + half * 16;

  v8f acc0 = {0,0,0,0,0,0,0,0}, acc1 = {0,0,0,0,0,0,0,0};
  v8f acc2 = {0,0,0,0,0,0,0,0}, acc3 = {0,0,0,0,0,0,0,0};
  v8f acc4 = {0,0,0,0,0,0,0,0}, acc5 = {0,0,0,0,0,0,0,0};
  v8f acc6 = {0,0,0,0,0,0,0,0}, acc7 = {0,0,0,0,0,0,0,0};

  for (int kk = 0; kk < K; kk += 32) {
    v16bf af;
    {
      const float* ap = arow + kk;                  // A layout: contiguous run at k+h*16 (matches B)
      float4 a0 = *(const float4*)(ap);
      float4 a1 = *(const float4*)(ap + 4);
      float4 a2 = *(const float4*)(ap + 8);
      float4 a3 = *(const float4*)(ap + 12);
      cvt2bf(a0.x, a0.y, af, 0);  cvt2bf(a0.z, a0.w, af, 2);
      cvt2bf(a1.x, a1.y, af, 4);  cvt2bf(a1.z, a1.w, af, 6);
      cvt2bf(a2.x, a2.y, af, 8);  cvt2bf(a2.z, a2.w, af, 10);
      cvt2bf(a3.x, a3.y, af, 12); cvt2bf(a3.z, a3.w, af, 14);
      __builtin_prefetch(arow + kk + 32, 0, 1);     // global_prefetch_b8
    }
    {
      v16bf b0 = load16bf(bp0 + kk);
      v16bf b1 = load16bf(bp1 + kk);
      v16bf b2 = load16bf(bp2 + kk);
      v16bf b3 = load16bf(bp3 + kk);
      acc0 = __builtin_amdgcn_wmma_f32_16x16x32_bf16(false, af, false, b0, (short)0, acc0, false, false);
      acc1 = __builtin_amdgcn_wmma_f32_16x16x32_bf16(false, af, false, b1, (short)0, acc1, false, false);
      acc2 = __builtin_amdgcn_wmma_f32_16x16x32_bf16(false, af, false, b2, (short)0, acc2, false, false);
      acc3 = __builtin_amdgcn_wmma_f32_16x16x32_bf16(false, af, false, b3, (short)0, acc3, false, false);
    }
    {
      v16bf b4 = load16bf(bp4 + kk);
      v16bf b5 = load16bf(bp5 + kk);
      v16bf b6 = load16bf(bp6 + kk);
      v16bf b7 = load16bf(bp7 + kk);
      acc4 = __builtin_amdgcn_wmma_f32_16x16x32_bf16(false, af, false, b4, (short)0, acc4, false, false);
      acc5 = __builtin_amdgcn_wmma_f32_16x16x32_bf16(false, af, false, b5, (short)0, acc5, false, false);
      acc6 = __builtin_amdgcn_wmma_f32_16x16x32_bf16(false, af, false, b6, (short)0, acc6, false, false);
      acc7 = __builtin_amdgcn_wmma_f32_16x16x32_bf16(false, af, false, b7, (short)0, acc7, false, false);
    }
  }

  const int ncol = nt + lanelow;
  float bi[8];
  #pragma unroll
  for (int j = 0; j < 8; j++) bi[j] = bias[ncol + j * 16];
  #pragma unroll
  for (int v = 0; v < 8; v++) {
    int m = mt * 16 + half * 8 + v;                 // ISA C/D layout: lane half picks M 0-7 / 8-15
    if (m < M) {
      float* crow = Cm + (size_t)m * N + ncol;
      float x0 = acc0[v] + bi[0], x1 = acc1[v] + bi[1], x2 = acc2[v] + bi[2], x3 = acc3[v] + bi[3];
      float x4 = acc4[v] + bi[4], x5 = acc5[v] + bi[5], x6 = acc6[v] + bi[6], x7 = acc7[v] + bi[7];
      if (relu) {
        x0=fmaxf(x0,0.f); x1=fmaxf(x1,0.f); x2=fmaxf(x2,0.f); x3=fmaxf(x3,0.f);
        x4=fmaxf(x4,0.f); x5=fmaxf(x5,0.f); x6=fmaxf(x6,0.f); x7=fmaxf(x7,0.f);
      }
      crow[0]  = x0; crow[16] = x1; crow[32] = x2; crow[48]  = x3;
      crow[64] = x4; crow[80] = x5; crow[96] = x6; crow[112] = x7;
    }
  }
}

// ---------------- weight transpose + bf16 convert: W[K,N](f32) -> Wt[N,Kp](bf16) ----------------
__global__ void k_prep_wt(const float* __restrict__ W, unsigned short* __restrict__ Wt,
                          int K, int N, int Kp) {
  int idx = blockIdx.x * 256 + threadIdx.x;
  if (idx >= N * Kp) return;
  int n = idx / Kp, k = idx - n * Kp;
  float v = (k < K) ? W[(size_t)k * N + n] : 0.0f;
  unsigned u = __float_as_uint(v);
  u += 0x7fffu + ((u >> 16) & 1u);
  Wt[idx] = (unsigned short)(u >> 16);
}

// ---------------- pad rows: X[M,K] -> Xp[M,Kp] (zero pad) ----------------
__global__ void k_pad_x(const float* __restrict__ X, float* __restrict__ Xp, int M, int K, int Kp) {
  int idx = blockIdx.x * 256 + threadIdx.x;
  if (idx >= M * Kp) return;
  int m = idx / Kp, k = idx - m * Kp;
  Xp[idx] = (k < K) ? X[(size_t)m * K + k] : 0.0f;
}

__global__ void k_fill(float* p, float v, int n) {
  int i = blockIdx.x * 256 + threadIdx.x;
  if (i < n) p[i] = v;
}

// ---------------- edge builder ----------------
__global__ void k_edge_check(const float* __restrict__ ea, int* flag, int E) {
  int e = blockIdx.x * 256 + threadIdx.x;
  if (e >= E) return;
  float second = ea[e * 2 + 1];
  if (!(second >= 0.0f && second <= 1.0f)) atomicOr(flag, 1);
}
__global__ void k_edge_build(const float* __restrict__ ea, const int* __restrict__ flag,
                             const float* __restrict__ lamp, float* __restrict__ feat, int E) {
  int e = blockIdx.x * 256 + threadIdx.x;
  if (e >= E) return;
  float s      = fminf(fmaxf(ea[e * 2], -1.0f), 1.0f);
  float second = ea[e * 2 + 1];
  float lam    = fmaxf(lamp[0], 1e-6f);
  float decay, dt;
  if (*flag == 0) {                                  // looks_like_decay
    decay = fminf(fmaxf(second, 0.0f), 1.0f);
    dt    = -logf(fmaxf(decay, 1e-6f)) / lam;
  } else {
    dt    = fmaxf(second, 0.0f);
    decay = expf(-lam * dt);
  }
  float t = log1pf(fmaxf(dt, 0.0f)) / 30.0f;
  float* o = feat + (size_t)e * 10;
  o[0] = s; o[1] = decay;
  #pragma unroll
  for (int k = 1; k <= 7; k++) o[1 + k] = sinf(t * (float)k);
  o[9] = t;
}

// ---------------- edge logits (xe computed on the fly, We/att in LDS) ----------------
__global__ __launch_bounds__(256) void k_edge_logits(
    const float* __restrict__ XL, const float* __restrict__ XR, const int* __restrict__ ei,
    const float* __restrict__ efeat, const float* __restrict__ We, const float* __restrict__ att,
    float* __restrict__ logits, float* __restrict__ dmax, int E)
{
  __shared__ float sWe[10 * 256];
  __shared__ float sAtt[256];
  for (int i = threadIdx.x; i < 2560; i += 256) sWe[i] = We[i];
  if (threadIdx.x < 256) sAtt[threadIdx.x] = att[threadIdx.x];
  __syncthreads();
  int wid = threadIdx.x >> 5, lane = threadIdx.x & 31;
  int e = blockIdx.x * 8 + wid;
  if (e >= E) return;
  int src = ei[e], dst = ei[E + e];
  float f[10];
  #pragma unroll
  for (int k = 0; k < 10; k++) f[k] = efeat[(size_t)e * 10 + k];
  const float* xl = XL + (size_t)src * 256;
  const float* xr = XR + (size_t)dst * 256;
  float part = 0.0f;
  int c0 = lane * 8;                                  // lane covers 8 channels of one head
  #pragma unroll
  for (int i = 0; i < 8; i++) {
    int c = c0 + i;
    float xe = 0.0f;
    #pragma unroll
    for (int k = 0; k < 10; k++) xe += f[k] * sWe[k * 256 + c];
    float m = xl[c] + xr[c] + xe;
    m = (m > 0.0f) ? m : 0.2f * m;                    // leaky_relu 0.2
    part += m * sAtt[c];
  }
  part += __shfl_xor(part, 1, 32);
  part += __shfl_xor(part, 2, 32);
  if ((lane & 3) == 0) {
    int h = lane >> 2;
    logits[(size_t)e * 8 + h] = part;
    atomicMaxF(&dmax[(size_t)dst * 8 + h], part);
  }
}

// ---------------- exp(logit - max) + segment sum ----------------
__global__ void k_edge_softmax(const int* __restrict__ ei, float* __restrict__ logits,
                               const float* __restrict__ dmax, float* __restrict__ z, int E) {
  int idx = blockIdx.x * 256 + threadIdx.x;
  if (idx >= E * 8) return;
  int e = idx >> 3, h = idx & 7;
  int dst = ei[E + e];
  float m = dmax[(size_t)dst * 8 + h];
  if (!(m > -1e37f)) m = 0.0f;                        // where(isfinite(m), m, 0)
  float v = expf(logits[idx] - m);
  logits[idx] = v;
  atomicAdd(&z[(size_t)dst * 8 + h], v);
}

// ---------------- aggregate: AGG[dst] += XL[src] * alpha ----------------
__global__ void k_edge_aggregate(const int* __restrict__ ei, const float* __restrict__ logits,
                                 const float* __restrict__ z, const float* __restrict__ XL,
                                 float* __restrict__ AGG, int E) {
  long long idx = (long long)blockIdx.x * 256 + threadIdx.x;
  if (idx >= (long long)E * 64) return;
  int e = (int)(idx >> 6), g = (int)(idx & 63);
  int src = ei[e], dst = ei[E + e];
  int h = g >> 3;
  float a = logits[(size_t)e * 8 + h] / (z[(size_t)dst * 8 + h] + 1e-16f);
  float4 v = *(const float4*)(XL + (size_t)src * 256 + g * 4);
  float* o = AGG + (size_t)dst * 256 + g * 4;
  atomicAdd(o + 0, v.x * a);
  atomicAdd(o + 1, v.y * a);
  atomicAdd(o + 2, v.z * a);
  atomicAdd(o + 3, v.w * a);
}

// ---------------- (x + bias) -> layernorm(g,b), one wave per row ----------------
__global__ __launch_bounds__(256) void k_bias_ln(const float* __restrict__ X, const float* __restrict__ bias,
                                                 const float* __restrict__ g, const float* __restrict__ b,
                                                 float* __restrict__ Y, int rows) {
  int wid = threadIdx.x >> 5, lane = threadIdx.x & 31;
  int row = blockIdx.x * 8 + wid;
  if (row >= rows) return;
  const float* x = X + (size_t)row * 256;
  float vals[8]; float s = 0.0f;
  #pragma unroll
  for (int i = 0; i < 8; i++) { int c = lane * 8 + i; vals[i] = x[c] + bias[c]; s += vals[i]; }
  for (int off = 16; off >= 1; off >>= 1) s += __shfl_xor(s, off, 32);
  float mu = s * (1.0f / 256.0f);
  float vs = 0.0f;
  #pragma unroll
  for (int i = 0; i < 8; i++) { float d = vals[i] - mu; vs += d * d; }
  for (int off = 16; off >= 1; off >>= 1) vs += __shfl_xor(vs, off, 32);
  float inv = rsqrtf(vs * (1.0f / 256.0f) + 1e-5f);
  float* y = Y + (size_t)row * 256;
  #pragma unroll
  for (int i = 0; i < 8; i++) { int c = lane * 8 + i; y[c] = (vals[i] - mu) * inv * g[c] + b[c]; }
}

// ---------------- pooling ----------------
__global__ void k_pool_fact(const float* __restrict__ Hf, const int* __restrict__ batch,
                            float* __restrict__ sum, float* __restrict__ cnt, int Nn) {
  int idx = blockIdx.x * 256 + threadIdx.x;
  if (idx >= Nn * 64) return;
  int node = idx >> 6, gch = idx & 63;
  int bkt = batch[node];
  float4 v = *(const float4*)(Hf + (size_t)node * 256 + gch * 4);
  float* o = sum + (size_t)bkt * 256 + gch * 4;
  atomicAdd(o + 0, v.x); atomicAdd(o + 1, v.y); atomicAdd(o + 2, v.z); atomicAdd(o + 3, v.w);
  if (gch == 0) atomicAdd(&cnt[bkt], 1.0f);
}
__global__ void k_pool_company(const float* __restrict__ Hc, const int* __restrict__ batch,
                               const unsigned char* __restrict__ mask,
                               float* __restrict__ msum, float* __restrict__ asum,
                               float* __restrict__ mcnt, float* __restrict__ acnt, int Nn) {
  int idx = blockIdx.x * 256 + threadIdx.x;
  if (idx >= Nn * 64) return;
  int node = idx >> 6, gch = idx & 63;
  int bkt = batch[node];
  bool w = mask[node] != 0;
  float4 v = *(const float4*)(Hc + (size_t)node * 256 + gch * 4);
  float* oa = asum + (size_t)bkt * 256 + gch * 4;
  atomicAdd(oa + 0, v.x); atomicAdd(oa + 1, v.y); atomicAdd(oa + 2, v.z); atomicAdd(oa + 3, v.w);
  if (w) {
    float* om = msum + (size_t)bkt * 256 + gch * 4;
    atomicAdd(om + 0, v.x); atomicAdd(om + 1, v.y); atomicAdd(om + 2, v.z); atomicAdd(om + 3, v.w);
  }
  if (gch == 0) { atomicAdd(&acnt[bkt], 1.0f); if (w) atomicAdd(&mcnt[bkt], 1.0f); }
}
__global__ void k_pool_finalize(const float* fsum, const float* fcnt,
                                const float* msum, const float* mcnt,
                                const float* asum, const float* acnt,
                                float* both, int Bn) {
  int idx = blockIdx.x * 256 + threadIdx.x;
  if (idx >= Bn * 256) return;
  int b = idx >> 8, c = idx & 255;
  float fp = fsum[idx] / fmaxf(fcnt[b], 1.0f);
  float mc = mcnt[b];
  float cp = (mc > 1e-6f) ? (msum[idx] / fmaxf(mc, 1.0f))
                          : (asum[idx] / fmaxf(acnt[b], 1.0f));
  both[(size_t)b * 512 + c]       = fp;
  both[(size_t)b * 512 + 256 + c] = cp;
}
__global__ __launch_bounds__(256) void k_gate(const float* __restrict__ both,
                                              const float* __restrict__ gW, const float* __restrict__ gb,
                                              float* __restrict__ pooled, int Bn) {
  int wid = threadIdx.x >> 5, lane = threadIdx.x & 31;
  int b = blockIdx.x * 8 + wid;
  if (b >= Bn) return;
  const float* r = both + (size_t)b * 512;
  float s = 0.0f;
  for (int i = lane; i < 512; i += 32) s += r[i] * gW[i];
  for (int off = 16; off >= 1; off >>= 1) s += __shfl_xor(s, off, 32);
  float alpha = 1.0f / (1.0f + expf(-(s + gb[0])));
  for (int c = lane; c < 256; c += 32)
    pooled[(size_t)b * 256 + c] = alpha * r[c] + (1.0f - alpha) * r[256 + c];
}
__global__ __launch_bounds__(256) void k_cls_out(const float* __restrict__ Hb,
                                                 const float* __restrict__ W2, const float* __restrict__ b2,
                                                 float* __restrict__ out, int Bn) {
  int wid = threadIdx.x >> 5, lane = threadIdx.x & 31;
  int b = blockIdx.x * 8 + wid;
  if (b >= Bn) return;
  const float* r = Hb + (size_t)b * 256;
  float s = 0.0f;
  for (int i = lane; i < 256; i += 32) s += r[i] * W2[i];
  for (int off = 16; off >= 1; off >>= 1) s += __shfl_xor(s, off, 32);
  if (lane == 0) out[b] = s + b2[0];
}

// ---------------- host orchestration ----------------
static inline int ceil_div(long long a, long long b) { return (int)((a + b - 1) / b); }

static void launch_gemm(const float* A, const unsigned short* Bt, const float* bias,
                        float* Cm, int M, int N, int K, int relu, hipStream_t s) {
  int waves = ceil_div(M, 16) * (N / 128);
  k_gemm_bf16<<<ceil_div(waves, 4), 128, 0, s>>>(A, Bt, bias, Cm, M, N, K, relu);
}
static void launch_prep_wt(const float* W, unsigned short* Wt, int K, int N, int Kp, hipStream_t s) {
  k_prep_wt<<<ceil_div((long long)N * Kp, 256), 256, 0, s>>>(W, Wt, K, N, Kp);
}

extern "C" void kernel_launch(void* const* d_in, const int* in_sizes, int n_in,
                              void* d_out, int out_size, void* d_ws, size_t ws_size,
                              hipStream_t stream) {
  (void)in_sizes; (void)n_in; (void)ws_size; (void)out_size;
  // ---- inputs (setup_inputs dict order; params flattened in dict insertion order) ----
  const float* x_fact    = (const float*)d_in[0];
  const float* x_company = (const float*)d_in[1];
  const int*   ei_fc     = (const int*)d_in[2];
  const float* ea_fc     = (const float*)d_in[3];
  const int*   ei_cf     = (const int*)d_in[4];
  const float* ea_cf     = (const float*)d_in[5];
  const int*   batch_f   = (const int*)d_in[6];
  const int*   batch_c   = (const int*)d_in[7];
  const unsigned char* pmask = (const unsigned char*)d_in[8];
  const float* lam       = (const float*)d_in[9];
  const float* inf_W     = (const float*)d_in[10];
  const float* inf_b     = (const float*)d_in[11];
  const float* inc_W     = (const float*)d_in[12];
  const float* inc_b     = (const float*)d_in[13];
  // layers at 14 + l*18 : fc{Wl,bl,Wr,br,We,att,bias}, cf{...}, ln_fact{g,b}, ln_company{g,b}
  const float* gate_W = (const float*)d_in[50];
  const float* gate_b = (const float*)d_in[51];
  const float* cls_W1 = (const float*)d_in[52];
  const float* cls_b1 = (const float*)d_in[53];
  const float* cls_W2 = (const float*)d_in[54];
  const float* cls_b2 = (const float*)d_in[55];
  float* out = (float*)d_out;

  // ---- workspace bump allocator ----
  char* wp = (char*)d_ws;
  auto alloc = [&](size_t bytes) -> char* {
    char* r = wp; wp += (bytes + 255) & ~(size_t)255; return r;
  };
  float* eaFC   = (float*)alloc((size_t)kE * 10 * 4);
  float* eaCF   = (float*)alloc((size_t)kE * 10 * 4);
  int*   flagFC = (int*)alloc(256);
  int*   flagCF = (int*)alloc(256);
  float* xcp    = (float*)alloc((size_t)kNC * 32 * 4);
  unsigned short* wt_inf = (unsigned short*)alloc((size_t)256 * 768 * 2);
  unsigned short* wt_inc = (unsigned short*)alloc((size_t)256 * 32 * 2);
  unsigned short* wtL[2][4];
  for (int l = 0; l < 2; l++)
    for (int j = 0; j < 4; j++)
      wtL[l][j] = (unsigned short*)alloc((size_t)256 * 256 * 2);
  unsigned short* wt_cls = (unsigned short*)alloc((size_t)256 * 256 * 2);
  float* hf   = (float*)alloc((size_t)kNF * 256 * 4);
  float* hc   = (float*)alloc((size_t)kNC * 256 * 4);
  float* XLf  = (float*)alloc((size_t)kNF * 256 * 4);
  float* XRf  = (float*)alloc((size_t)kNF * 256 * 4);
  float* XLc  = (float*)alloc((size_t)kNC * 256 * 4);
  float* XRc  = (float*)alloc((size_t)kNC * 256 * 4);
  float* AGGf = (float*)alloc((size_t)kNF * 256 * 4);
  float* AGGc = (float*)alloc((size_t)kNC * 256 * 4);
  float* logits = (float*)alloc((size_t)kE * 8 * 4);
  float* dmax   = (float*)alloc((size_t)kNF * 8 * 4);
  float* zb     = (float*)alloc((size_t)kNF * 8 * 4);
  float* fsum = (float*)alloc((size_t)kB * 256 * 4);
  float* fcnt = (float*)alloc((size_t)kB * 4);
  float* msum = (float*)alloc((size_t)kB * 256 * 4);
  float* mcnt = (float*)alloc((size_t)kB * 4);
  float* asum = (float*)alloc((size_t)kB * 256 * 4);
  float* acnt = (float*)alloc((size_t)kB * 4);
  float* both   = (float*)alloc((size_t)kB * 512 * 4);
  float* pooled = (float*)alloc((size_t)kB * 256 * 4);
  float* hcls   = (float*)alloc((size_t)kB * 256 * 4);

  // ---- edge features (layer-independent) ----
  hipMemsetAsync(flagFC, 0, 4, stream);
  hipMemsetAsync(flagCF, 0, 4, stream);
  k_edge_check<<<ceil_div(kE, 256), 256, 0, stream>>>(ea_fc, flagFC, kE);
  k_edge_check<<<ceil_div(kE, 256), 256, 0, stream>>>(ea_cf, flagCF, kE);
  k_edge_build<<<ceil_div(kE, 256), 256, 0, stream>>>(ea_fc, flagFC, lam, eaFC, kE);
  k_edge_build<<<ceil_div(kE, 256), 256, 0, stream>>>(ea_cf, flagCF, lam, eaCF, kE);

  // ---- weight prep (transpose + bf16) ----
  launch_prep_wt(inf_W, wt_inf, 768, 256, 768, stream);
  launch_prep_wt(inc_W, wt_inc, 27, 256, 32, stream);
  for (int l = 0; l < 2; l++) {
    int base = 14 + l * 18;
    launch_prep_wt((const float*)d_in[base + 0], wtL[l][0], 256, 256, 256, stream); // fc.Wl
    launch_prep_wt((const float*)d_in[base + 2], wtL[l][1], 256, 256, 256, stream); // fc.Wr
    launch_prep_wt((const float*)d_in[base + 7], wtL[l][2], 256, 256, 256, stream); // cf.Wl
    launch_prep_wt((const float*)d_in[base + 9], wtL[l][3], 256, 256, 256, stream); // cf.Wr
  }
  launch_prep_wt(cls_W1, wt_cls, 256, 256, 256, stream);
  k_pad_x<<<ceil_div((long long)kNC * 32, 256), 256, 0, stream>>>(x_company, xcp, kNC, 27, 32);

  // ---- input projections (WMMA) ----
  launch_gemm(x_fact, wt_inf, inf_b, hf, kNF, 256, 768, 1, stream);
  launch_gemm(xcp,    wt_inc, inc_b, hc, kNC, 256, 32,  1, stream);

  // ---- GAT layers ----
  auto run_gat = [&](const int* ei, const float* efeat, const float* XL, const float* XR,
                     const float* We, const float* att, float* AGG, int n_dst) {
    hipMemsetAsync(AGG, 0, (size_t)n_dst * 256 * 4, stream);
    hipMemsetAsync(zb, 0, (size_t)n_dst * 8 * 4, stream);
    k_fill<<<ceil_div((long long)n_dst * 8, 256), 256, 0, stream>>>(dmax, -INFINITY, n_dst * 8);
    k_edge_logits<<<ceil_div(kE, 8), 256, 0, stream>>>(XL, XR, ei, efeat, We, att, logits, dmax, kE);
    k_edge_softmax<<<ceil_div((long long)kE * 8, 256), 256, 0, stream>>>(ei, logits, dmax, zb, kE);
    k_edge_aggregate<<<ceil_div((long long)kE * 64, 256), 256, 0, stream>>>(ei, logits, zb, XL, AGG, kE);
  };

  for (int l = 0; l < 2; l++) {
    int base = 14 + l * 18;
    const float* fc_bl   = (const float*)d_in[base + 1];
    const float* fc_br   = (const float*)d_in[base + 3];
    const float* fc_We   = (const float*)d_in[base + 4];
    const float* fc_att  = (const float*)d_in[base + 5];
    const float* fc_bias = (const float*)d_in[base + 6];
    const float* cf_bl   = (const float*)d_in[base + 8];
    const float* cf_br   = (const float*)d_in[base + 10];
    const float* cf_We   = (const float*)d_in[base + 11];
    const float* cf_att  = (const float*)d_in[base + 12];
    const float* cf_bias = (const float*)d_in[base + 13];
    const float* lnf_g   = (const float*)d_in[base + 14];
    const float* lnf_b   = (const float*)d_in[base + 15];
    const float* lnc_g   = (const float*)d_in[base + 16];
    const float* lnc_b   = (const float*)d_in[base + 17];

    // projections from OLD hf/hc (WMMA GEMMs)
    launch_gemm(hf, wtL[l][0], fc_bl, XLf, kNF, 256, 256, 0, stream); // fc: xl = hf@Wl
    launch_gemm(hc, wtL[l][1], fc_br, XRc, kNC, 256, 256, 0, stream); // fc: xr = hc@Wr
    launch_gemm(hc, wtL[l][2], cf_bl, XLc, kNC, 256, 256, 0, stream); // cf: xl = hc@Wl
    launch_gemm(hf, wtL[l][3], cf_br, XRf, kNF, 256, 256, 0, stream); // cf: xr = hf@Wr

    run_gat(ei_fc, eaFC, XLf, XRc, fc_We, fc_att, AGGc, kNC);         // new_c
    run_gat(ei_cf, eaCF, XLc, XRf, cf_We, cf_att, AGGf, kNF);         // new_f

    k_bias_ln<<<ceil_div(kNC, 8), 256, 0, stream>>>(AGGc, fc_bias, lnc_g, lnc_b, hc, kNC);
    k_bias_ln<<<ceil_div(kNF, 8), 256, 0, stream>>>(AGGf, cf_bias, lnf_g, lnf_b, hf, kNF);
  }

  // ---- pooling + gate + classifier ----
  hipMemsetAsync(fsum, 0, (size_t)kB * 256 * 4, stream);
  hipMemsetAsync(fcnt, 0, (size_t)kB * 4, stream);
  hipMemsetAsync(msum, 0, (size_t)kB * 256 * 4, stream);
  hipMemsetAsync(mcnt, 0, (size_t)kB * 4, stream);
  hipMemsetAsync(asum, 0, (size_t)kB * 256 * 4, stream);
  hipMemsetAsync(acnt, 0, (size_t)kB * 4, stream);
  k_pool_fact<<<ceil_div((long long)kNF * 64, 256), 256, 0, stream>>>(hf, batch_f, fsum, fcnt, kNF);
  k_pool_company<<<ceil_div((long long)kNC * 64, 256), 256, 0, stream>>>(hc, batch_c, pmask,
                                                                          msum, asum, mcnt, acnt, kNC);
  k_pool_finalize<<<ceil_div((long long)kB * 256, 256), 256, 0, stream>>>(fsum, fcnt, msum, mcnt,
                                                                          asum, acnt, both, kB);
  k_gate<<<ceil_div(kB, 8), 256, 0, stream>>>(both, gate_W, gate_b, pooled, kB);
  launch_gemm(pooled, wt_cls, cls_b1, hcls, kB, 256, 256, 1, stream);
  k_cls_out<<<ceil_div(kB, 8), 256, 0, stream>>>(hcls, cls_W2, cls_b2, out, kB);
}
